// CausalSelfAttention_25383256719396
// MI455X (gfx1250) — compile-verified, matrix operand order checked by reading, not run on hardware
//
#include <hip/hip_runtime.h>
#include <hip/hip_bf16.h>

typedef unsigned short u16;
typedef unsigned int   u32;

typedef __attribute__((ext_vector_type(16))) __bf16 v16bf;
typedef __attribute__((ext_vector_type(8)))  __bf16 v8bf;
typedef __attribute__((ext_vector_type(8)))  float  v8f;

#define SEQ   2048
#define DIM   2048
#define DH    64
#define NH    32
#define QKVN  6144

// ---- helpers -------------------------------------------------------------

__device__ inline u16 f2bf(float f) {
  u32 u = __float_as_uint(f);
  u += 0x7FFFu + ((u >> 16) & 1u);      // round-to-nearest-even
  return (u16)(u >> 16);
}

// Load one 16x32 bf16 WMMA A/B fragment from a row-major [rows x ld] buffer.
// ISA layout: lane r (0..15) holds row row0+r; half=lane>>4 selects K phase.
// Elements 0..7  : K = k0 + half*8 .. +7   (one contiguous 16B chunk)
// Elements 8..15 : K = k0 + 16 + half*8 .. (second contiguous 16B chunk)
__device__ inline v16bf load_frag(const u16* p, int ld, int row0, int k0, int lane) {
  int r = lane & 15, h = (lane >> 4) & 1;
  const u16* b0 = p + (size_t)(row0 + r) * ld + k0 + h * 8;
  v8bf lo = *(const v8bf*)b0;
  v8bf hi = *(const v8bf*)(b0 + 16);
  return __builtin_shufflevector(lo, hi, 0,1,2,3,4,5,6,7,8,9,10,11,12,13,14,15);
}

// ---- precision staging ---------------------------------------------------

__global__ void __launch_bounds__(256) cvt_kernel(const float* __restrict__ in,
                                                  u16* __restrict__ out, int n) {
  int i = blockIdx.x * 256 + threadIdx.x;
  if (i < n) out[i] = f2bf(in[i]);
}

// in: [rows x cols] row-major f32 -> out: [cols x rows] row-major bf16 (B^T)
__global__ void __launch_bounds__(256) tcvt_kernel(const float* __restrict__ in,
                                                   u16* __restrict__ out,
                                                   int rows, int cols) {
  int i = blockIdx.x * 256 + threadIdx.x;
  if (i < rows * cols) {
    int k = i / cols, n = i - k * cols;
    out[(size_t)n * rows + k] = f2bf(in[i]);
  }
}

// ---- QKV GEMM: [2048x2048] @ [2048x6144] -> scattered q/k/vT bf16 --------

__global__ void __launch_bounds__(256) qkv_gemm_kernel(const u16* __restrict__ xb,
                                                       const u16* __restrict__ wT,
                                                       u16* __restrict__ qb,
                                                       u16* __restrict__ kb,
                                                       u16* __restrict__ vT) {
  int lane = threadIdx.x & 31;
  int w    = blockIdx.x * 8 + (threadIdx.x >> 5);
  int m0   = (w & 31) * 64;          // 32 M tiles
  int n0   = (w >> 5) * 64;          // 96 N tiles
  v8f acc[4][4] = {};
  for (int k0 = 0; k0 < DIM; k0 += 32) {
    // Issue all 16 b128 loads up front so the scheduler can stagger
    // s_wait_loadcnt against the 16 WMMAs instead of hard 0x0 waits.
    v16bf a[4], b[4];
#pragma unroll
    for (int i = 0; i < 4; i++) a[i] = load_frag(xb, DIM, m0 + 16 * i, k0, lane);
#pragma unroll
    for (int j = 0; j < 4; j++) b[j] = load_frag(wT, DIM, n0 + 16 * j, k0, lane);
#pragma unroll
    for (int j = 0; j < 4; j++)
#pragma unroll
      for (int i = 0; i < 4; i++)
        acc[i][j] = __builtin_amdgcn_wmma_f32_16x16x32_bf16(
            false, a[i], false, b[j], (short)0, acc[i][j], false, false);
  }
  // Epilogue: fully unrolled so acc[][] stays in VGPRs (no scratch spill).
  int r = lane & 15, h = (lane >> 4) & 1;
#pragma unroll
  for (int j = 0; j < 4; j++) {
    int N16  = n0 + 16 * j;
    int sec  = N16 >> 11;            // 0=Q 1=K 2=V (uniform per tile)
    int cn   = N16 & 2047;
    int head = cn >> 6;
    int d    = (cn & 63) + r;
    u16* qkbase = (sec == 0 ? qb : kb) + (size_t)head * SEQ * DH + d;
    u16* vbase  = vT + ((size_t)head * DH + d) * SEQ;
#pragma unroll
    for (int i = 0; i < 4; i++) {
      int sbase = m0 + 16 * i + 8 * h;
#pragma unroll
      for (int e = 0; e < 8; e++) {
        int s   = sbase + e;
        float v = acc[i][j][e];
        if (sec == 0)      qkbase[(size_t)s * DH] = f2bf(v * 0.125f);
        else if (sec == 1) qkbase[(size_t)s * DH] = f2bf(v);
        else               vbase[s]               = f2bf(v);
      }
    }
  }
}

// ---- Flash attention: one wave = 16 query rows of one head ---------------

__global__ void __launch_bounds__(256) attn_kernel(const u16* __restrict__ qb,
                                                   const u16* __restrict__ kb,
                                                   const u16* __restrict__ vT,
                                                   u16* __restrict__ attnb) {
  __shared__ u16 pbuf[8][16 * 32];   // per-wave P-tile bounce buffer (8KB)
  int lane = threadIdx.x & 31;
  int wv   = threadIdx.x >> 5;
  int h    = blockIdx.y;
  int q0   = (blockIdx.x * 8 + wv) * 16;
  int r    = lane & 15, hf = (lane >> 4) & 1;
  const u16* qh = qb + (size_t)h * SEQ * DH;
  const u16* kh = kb + (size_t)h * SEQ * DH;
  const u16* vh = vT + (size_t)h * DH * SEQ;
  u16* pw = &pbuf[wv][0];

  v16bf qf0 = load_frag(qh, DH, q0, 0,  lane);
  v16bf qf1 = load_frag(qh, DH, q0, 32, lane);

  v8f o[4] = {};
  float mi[8], li[8];
#pragma unroll
  for (int e = 0; e < 8; e++) { mi[e] = -3e30f; li[e] = 0.f; }

  for (int t0 = 0; t0 <= q0 + 15; t0 += 32) {
    // S = Q @ K^T for a 16x32 key block (two 16x16 N tiles)
    v8f s0 = {}, s1 = {};
    {
      v16bf b00 = load_frag(kh, DH, t0,      0,  lane);
      v16bf b01 = load_frag(kh, DH, t0,      32, lane);
      v16bf b10 = load_frag(kh, DH, t0 + 16, 0,  lane);
      v16bf b11 = load_frag(kh, DH, t0 + 16, 32, lane);
      s0 = __builtin_amdgcn_wmma_f32_16x16x32_bf16(false, qf0, false, b00, (short)0, s0, false, false);
      s0 = __builtin_amdgcn_wmma_f32_16x16x32_bf16(false, qf1, false, b01, (short)0, s0, false, false);
      s1 = __builtin_amdgcn_wmma_f32_16x16x32_bf16(false, qf0, false, b10, (short)0, s1, false, false);
      s1 = __builtin_amdgcn_wmma_f32_16x16x32_bf16(false, qf1, false, b11, (short)0, s1, false, false);
    }
    // online softmax (row = 16-lane half-group; e indexes 8 rows per lane)
    float alpha[8];
#pragma unroll
    for (int e = 0; e < 8; e++) {
      int row = q0 + e + 8 * hf;
      int c0  = t0 + r;
      float a0 = (c0      <= row) ? s0[e] : -1e30f;
      float a1 = (c0 + 16 <= row) ? s1[e] : -1e30f;
      float mx = fmaxf(a0, a1);
#pragma unroll
      for (int m = 1; m < 16; m <<= 1) mx = fmaxf(mx, __shfl_xor(mx, m, 32));
      float mn = fmaxf(mi[e], mx);
      float al = __expf(mi[e] - mn);
      float p0 = __expf(a0 - mn);
      float p1 = __expf(a1 - mn);
      float sm = p0 + p1;
#pragma unroll
      for (int m = 1; m < 16; m <<= 1) sm += __shfl_xor(sm, m, 32);
      li[e]    = li[e] * al + sm;
      mi[e]    = mn;
      alpha[e] = al;
      int prow = e + 8 * hf;
      pw[prow * 32 + r]      = f2bf(p0);
      pw[prow * 32 + 16 + r] = f2bf(p1);
    }
    // C-layout -> A-layout reshape via same-wave LDS (in-order DS pipe)
    asm volatile("s_wait_dscnt 0" ::: "memory");
    v16bf pf = load_frag(pw, 32, 0, 0, lane);
    // O = O*alpha + P @ V  (V^T rows = d, contiguous t)
#pragma unroll
    for (int j = 0; j < 4; j++) {
      v8f c = {};
#pragma unroll
      for (int e = 0; e < 8; e++) c[e] = o[j][e] * alpha[e];
      v16bf vb = load_frag(vh, SEQ, j * 16, t0, lane);
      o[j] = __builtin_amdgcn_wmma_f32_16x16x32_bf16(false, pf, false, vb, (short)0, c, false, false);
    }
  }
#pragma unroll
  for (int j = 0; j < 4; j++) {
#pragma unroll
    for (int e = 0; e < 8; e++) {
      int s = q0 + e + 8 * hf;
      attnb[(size_t)s * DIM + h * DH + j * 16 + r] = f2bf(o[j][e] / li[e]);
    }
  }
}

// ---- Output projection: attn_bf16 @ Wout^T + bias -> f32 -----------------

__global__ void __launch_bounds__(256) out_gemm_kernel(const u16* __restrict__ ab,
                                                       const u16* __restrict__ wT,
                                                       const float* __restrict__ bias,
                                                       float* __restrict__ out) {
  int lane = threadIdx.x & 31;
  int w    = blockIdx.x * 8 + (threadIdx.x >> 5);
  int m0   = (w & 31) * 64;
  int n0   = (w >> 5) * 64;
  v8f acc[4][4] = {};
  for (int k0 = 0; k0 < DIM; k0 += 32) {
    v16bf a[4], b[4];
#pragma unroll
    for (int i = 0; i < 4; i++) a[i] = load_frag(ab, DIM, m0 + 16 * i, k0, lane);
#pragma unroll
    for (int j = 0; j < 4; j++) b[j] = load_frag(wT, DIM, n0 + 16 * j, k0, lane);
#pragma unroll
    for (int j = 0; j < 4; j++)
#pragma unroll
      for (int i = 0; i < 4; i++)
        acc[i][j] = __builtin_amdgcn_wmma_f32_16x16x32_bf16(
            false, a[i], false, b[j], (short)0, acc[i][j], false, false);
  }
  // Epilogue: fully unrolled, coalesced 64B row stores per (i,e).
  int r = lane & 15, h = (lane >> 4) & 1;
#pragma unroll
  for (int j = 0; j < 4; j++) {
    int n    = n0 + 16 * j + r;
    float bv = bias[n];
    float* obase = out + n;
#pragma unroll
    for (int i = 0; i < 4; i++) {
      int sbase = m0 + 16 * i + 8 * h;
#pragma unroll
      for (int e = 0; e < 8; e++) {
        obase[(size_t)(sbase + e) * DIM] = acc[i][j][e] + bv;
      }
    }
  }
}

// ---- launch --------------------------------------------------------------

extern "C" void kernel_launch(void* const* d_in, const int* in_sizes, int n_in,
                              void* d_out, int out_size, void* d_ws, size_t ws_size,
                              hipStream_t stream) {
  const float* x    = (const float*)d_in[0];
  const float* Wqkv = (const float*)d_in[1];
  const float* Wout = (const float*)d_in[2];
  const float* bias = (const float*)d_in[3];
  float* out = (float*)d_out;

  char* ws = (char*)d_ws;
  u16* xb    = (u16*)(ws + ((size_t)0  << 20));  //  8 MB  x bf16
  u16* wqkvT = (u16*)(ws + ((size_t)8  << 20));  // 24 MB  Wqkv^T bf16
  u16* woutT = (u16*)(ws + ((size_t)32 << 20));  //  8 MB  Wout^T bf16
  u16* qb    = (u16*)(ws + ((size_t)40 << 20));  //  8 MB  Q [h][s][64]
  u16* kb    = (u16*)(ws + ((size_t)48 << 20));  //  8 MB  K [h][s][64]
  u16* vT    = (u16*)(ws + ((size_t)56 << 20));  //  8 MB  V^T [h][64][s]
  u16* attnb = (u16*)(ws + ((size_t)64 << 20));  //  8 MB  attn [s][2048]

  cvt_kernel <<<(SEQ * DIM + 255) / 256, 256, 0, stream>>>(x, xb, SEQ * DIM);
  tcvt_kernel<<<(DIM * QKVN + 255) / 256, 256, 0, stream>>>(Wqkv, wqkvT, DIM, QKVN);
  tcvt_kernel<<<(DIM * DIM + 255) / 256, 256, 0, stream>>>(Wout, woutT, DIM, DIM);

  qkv_gemm_kernel<<<(32 * 96) / 8, 256, 0, stream>>>(xb, wqkvT, qb, kb, vT);
  attn_kernel    <<<dim3(SEQ / (16 * 8), NH), 256, 0, stream>>>(qb, kb, vT, attnb);
  out_gemm_kernel<<<(32 * 32) / 8, 256, 0, stream>>>(attnb, woutT, bias, out);
}
